// RPN_55293408968766
// MI455X (gfx1250) — matrix operand order repaired and checked by reading, not compiled, not run on hardware
//
#include <hip/hip_runtime.h>
#include <stdint.h>

typedef __attribute__((ext_vector_type(16))) _Float16 v16h;
typedef __attribute__((ext_vector_type(8)))  _Float16 v8h;
typedef __attribute__((ext_vector_type(8)))  float    v8f;
typedef __attribute__((ext_vector_type(4)))  float    v4f;

#define NBATCH    2
#define CIN       512
#define COUT      512
#define HW        64
#define SPATIAL   4096
#define NANCH     9
#define TOTANCH   36864
#define NPRE      6000
#define NPOST     300
#define SORTN     65536
#define NMS_THR   0.7f

// output offsets (in floats)
#define OFF_LOCS    0
#define OFF_SCORES  294912
#define OFF_ROIS    442368
#define OFF_RIDX    444768
#define OFF_ANCHOR  445368
#define OFF_VMASK   592824

// workspace offsets (bytes)
#define WS_XH   0ull          // f16 [2][64][64][512]      = 8,388,608
#define WS_WH   8388608ull    // f16 [512][9][512]         = 4,718,592
#define WS_HH   13107200ull   // f16 [2][4096][512]        = 8,388,608
#define WS_WC   21495808ull   // f16 [64][512]             = 65,536
#define WS_HO   21561344ull   // f32 [2][4096][64]         = 2,097,152
#define WS_BOX  23658496ull   // f32 [2][36864][4]         = 1,179,648
#define WS_KEYS 24838144ull   // u64 [2][65536]            = 1,048,576
#define WS_TOP  25886720ull   // f32 [2][6000][4]          = 192,000

__device__ __forceinline__ v16h ld16(const _Float16* p0, const _Float16* p1) {
  v8h lo = *(const v8h*)p0;
  v8h hi = *(const v8h*)p1;
  return __builtin_shufflevector(lo, hi, 0,1,2,3,4,5,6,7,8,9,10,11,12,13,14,15);
}

// ---- x (NCHW f32) -> xh (NHWC f16) ----
__global__ void k_cvt_x(const float* __restrict__ x, _Float16* __restrict__ xh) {
  int g = blockIdx.x * blockDim.x + threadIdx.x;
  if (g >= NBATCH * SPATIAL * CIN) return;
  int c    = g & (CIN - 1);
  int rest = g >> 9;               // n*4096 + p
  int p    = rest & (SPATIAL - 1);
  int n    = rest >> 12;
  xh[g] = (_Float16)x[((size_t)n * CIN + c) * SPATIAL + p];
}

// ---- conv1_w (O,I,3,3 f32) -> wh[o][kpos][c] f16 ----
__global__ void k_cvt_w(const float* __restrict__ w, _Float16* __restrict__ wh) {
  int g = blockIdx.x * blockDim.x + threadIdx.x;
  if (g >= COUT * 9 * CIN) return;
  int c = g & (CIN - 1);
  int r = g >> 9;                  // o*9 + kpos
  int kpos = r % 9;
  int o = r / 9;
  wh[g] = (_Float16)w[((size_t)o * CIN + c) * 9 + kpos];
}

// ---- combined head weights: rows 0-35 loc, 36-53 score, 54-63 zero ----
__global__ void k_cvt_wc(const float* __restrict__ loc_w, const float* __restrict__ score_w,
                         _Float16* __restrict__ wc) {
  int g = blockIdx.x * blockDim.x + threadIdx.x;
  if (g >= 64 * CIN) return;
  int c = g & (CIN - 1);
  int m = g >> 9;
  float v = 0.0f;
  if (m < 36)      v = loc_w[(size_t)m * CIN + c];
  else if (m < 54) v = score_w[(size_t)(m - 36) * CIN + c];
  wc[g] = (_Float16)v;
}

// ---- 3x3 conv as implicit GEMM with WMMA f16 (f32 accum), bias+ReLU, f16 NHWC out ----
// one wave: 16 out-channels x 64 spatial (one image row), 4 accumulators (A reused 4x).
// 3x3 taps fully unrolled: dy/dx are compile-time; out-of-image rows skipped uniformly
// (their B is all-zero => WMMA is a no-op); only the two true edge sub-tiles carry a
// per-lane guard (col 0 @ dx=-1, col 15 @ dx=+1).
__global__ void k_conv(const _Float16* __restrict__ xh, const _Float16* __restrict__ wh,
                       const float* __restrict__ bias, _Float16* __restrict__ hh) {
  const int wid  = blockIdx.x * (blockDim.x >> 5) + (threadIdx.x >> 5);
  const int lane = threadIdx.x & 31;
  const int col  = lane & 15;
  const int half = lane >> 4;
  const int n    = wid >> 11;      // 2048 waves per batch (32 oT * 64 rows)
  const int rem  = wid & 2047;
  const int oT   = rem >> 6;
  const int y    = rem & 63;
  const int o_row = oT * 16 + col; // A-matrix row (M = lane&15)

  v8f acc0 = {}, acc1 = {}, acc2 = {}, acc3 = {};
  const _Float16* wrow = wh + (size_t)o_row * (9 * CIN) + half * 8;
#pragma unroll
  for (int dy = -1; dy <= 1; ++dy) {
    const int yy = y + dy;
    if ((unsigned)yy >= (unsigned)HW) continue;   // wave-uniform skip (B would be 0)
    const long long rowbase = (long long)(n * HW + yy) * HW;
#pragma unroll
    for (int dx = -1; dx <= 1; ++dx) {
      const int kpos = (dy + 1) * 3 + (dx + 1);
      const _Float16* ap_base = wrow + kpos * CIN;
      const bool v0 = (col + dx) >= 0;            // false only for col==0, dx==-1
      const bool v3 = (48 + col + dx) < HW;       // false only for col==15, dx==+1
      const long long s0 = (rowbase + (col + dx)) * CIN + half * 8;
      const long long s1 = s0 + 16 * CIN;
      const long long s2 = s0 + 32 * CIN;
      const long long s3 = s0 + 48 * CIN;
      for (int cb = 0; cb < CIN; cb += 32) {
        v16h A  = ld16(ap_base + cb, ap_base + cb + 16);
        v16h B1 = ld16(xh + s1 + cb, xh + s1 + cb + 16);   // always in range
        v16h B2 = ld16(xh + s2 + cb, xh + s2 + cb + 16);   // always in range
        v16h B0 = {}, B3 = {};
        if (v0) B0 = ld16(xh + s0 + cb, xh + s0 + cb + 16);
        if (v3) B3 = ld16(xh + s3 + cb, xh + s3 + cb + 16);
        acc0 = __builtin_amdgcn_wmma_f32_16x16x32_f16(false, A, false, B0, (short)0, acc0, false, false);
        acc1 = __builtin_amdgcn_wmma_f32_16x16x32_f16(false, A, false, B1, (short)0, acc1, false, false);
        acc2 = __builtin_amdgcn_wmma_f32_16x16x32_f16(false, A, false, B2, (short)0, acc2, false, false);
        acc3 = __builtin_amdgcn_wmma_f32_16x16x32_f16(false, A, false, B3, (short)0, acc3, false, false);
      }
    }
  }
  const size_t sp_base = (size_t)n * SPATIAL + (size_t)y * HW;
#pragma unroll
  for (int t = 0; t < 4; ++t) {
    v8f av = (t == 0) ? acc0 : (t == 1) ? acc1 : (t == 2) ? acc2 : acc3;
    v8h hv;
#pragma unroll
    for (int r = 0; r < 8; ++r) {
      float v = av[r] + bias[oT * 16 + half * 8 + r];
      hv[r] = (_Float16)(v > 0.0f ? v : 0.0f);
    }
    *(v8h*)(hh + (sp_base + t * 16 + col) * COUT + oT * 16 + half * 8) = hv;
  }
}

// ---- 1x1 head convs via WMMA: 64 outputs x 16 spatial per wave, K=512 ----
__global__ void k_heads(const _Float16* __restrict__ hh, const _Float16* __restrict__ wc,
                        float* __restrict__ ho) {
  const int wid  = blockIdx.x * (blockDim.x >> 5) + (threadIdx.x >> 5);
  const int lane = threadIdx.x & 31;
  const int col  = lane & 15;
  const int half = lane >> 4;
  const int n  = wid >> 8;         // 256 waves per batch
  const int sT = wid & 255;
  const int p  = sT * 16 + col;
  v8f acc0 = {}, acc1 = {}, acc2 = {}, acc3 = {};
  const _Float16* brow = hh + ((size_t)n * SPATIAL + p) * COUT + half * 8;
  const _Float16* arow = wc + (size_t)col * COUT + half * 8;
  for (int cb = 0; cb < COUT; cb += 32) {
    v16h B = ld16(brow + cb, brow + cb + 16);
    const _Float16* a0 = arow + cb;
    v16h A0 = ld16(a0,                a0 + 16);
    v16h A1 = ld16(a0 + 16 * COUT,    a0 + 16 * COUT + 16);
    v16h A2 = ld16(a0 + 32 * COUT,    a0 + 32 * COUT + 16);
    v16h A3 = ld16(a0 + 48 * COUT,    a0 + 48 * COUT + 16);
    acc0 = __builtin_amdgcn_wmma_f32_16x16x32_f16(false, A0, false, B, (short)0, acc0, false, false);
    acc1 = __builtin_amdgcn_wmma_f32_16x16x32_f16(false, A1, false, B, (short)0, acc1, false, false);
    acc2 = __builtin_amdgcn_wmma_f32_16x16x32_f16(false, A2, false, B, (short)0, acc2, false, false);
    acc3 = __builtin_amdgcn_wmma_f32_16x16x32_f16(false, A3, false, B, (short)0, acc3, false, false);
  }
  float* base0 = ho + ((size_t)n * SPATIAL + p) * 64;
#pragma unroll
  for (int t = 0; t < 4; ++t) {
    v8f av = (t == 0) ? acc0 : (t == 1) ? acc1 : (t == 2) ? acc2 : acc3;
    float* dst = base0 + t * 16 + half * 8;
    v4f lo, hi;
#pragma unroll
    for (int r = 0; r < 4; ++r) { lo[r] = av[r]; hi[r] = av[r + 4]; }
    *(v4f*)dst       = lo;
    *(v4f*)(dst + 4) = hi;
  }
}

__global__ void k_init_keys(unsigned long long* __restrict__ keys) {
  int g = blockIdx.x * blockDim.x + threadIdx.x;
  if (g < NBATCH * SORTN) keys[g] = ~0ull;
}

// ---- per-anchor: outputs rpn_locs/rpn_scores/anchor, decode+clip boxes, fg score, sort key ----
__global__ void k_decode(const float* __restrict__ ho, const float* __restrict__ loc_b,
                         const float* __restrict__ score_b, const int* __restrict__ pimh,
                         const int* __restrict__ pimw, float* __restrict__ out,
                         float* __restrict__ boxws, unsigned long long* __restrict__ keys) {
  int g = blockIdx.x * blockDim.x + threadIdx.x;
  if (g >= NBATCH * TOTANCH) return;
  int n = g / TOTANCH;
  int f = g - n * TOTANCH;
  int p = f / NANCH;
  int a = f - p * NANCH;
  int y = p >> 6;
  int x = p & 63;
  const float* hb = ho + ((size_t)n * SPATIAL + p) * 64;
  float l0 = hb[a * 4 + 0] + loc_b[a * 4 + 0];
  float l1 = hb[a * 4 + 1] + loc_b[a * 4 + 1];
  float l2 = hb[a * 4 + 2] + loc_b[a * 4 + 2];
  float l3 = hb[a * 4 + 3] + loc_b[a * 4 + 3];
  float s0 = hb[36 + a * 2 + 0] + score_b[a * 2 + 0];
  float s1 = hb[36 + a * 2 + 1] + score_b[a * 2 + 1];

  size_t lbase = ((size_t)n * TOTANCH + f) * 4;
  out[OFF_LOCS + lbase + 0] = l0;
  out[OFF_LOCS + lbase + 1] = l1;
  out[OFF_LOCS + lbase + 2] = l2;
  out[OFF_LOCS + lbase + 3] = l3;
  size_t sbase = ((size_t)n * TOTANCH + f) * 2;
  out[OFF_SCORES + sbase + 0] = s0;
  out[OFF_SCORES + sbase + 1] = s1;

  const float ratios[3] = {0.5f, 1.0f, 2.0f};
  const float scales[3] = {8.0f, 16.0f, 32.0f};
  float rr = ratios[a / 3], ss = scales[a % 3];
  float hA = 16.0f * ss * sqrtf(rr);
  float wA = 16.0f * ss * sqrtf(1.0f / rr);
  float acy = (float)y * 16.0f + 8.0f;
  float acx = (float)x * 16.0f + 8.0f;
  if (n == 0) {
    out[OFF_ANCHOR + (size_t)f * 4 + 0] = acy - 0.5f * hA;
    out[OFF_ANCHOR + (size_t)f * 4 + 1] = acx - 0.5f * wA;
    out[OFF_ANCHOR + (size_t)f * 4 + 2] = acy + 0.5f * hA;
    out[OFF_ANCHOR + (size_t)f * 4 + 3] = acx + 0.5f * wA;
  }
  float imh = (float)pimh[0], imw = (float)pimw[0];
  float cy = l0 * hA + acy;
  float cx = l1 * wA + acx;
  float bh = expf(l2) * hA;
  float bw = expf(l3) * wA;
  float y1 = fminf(fmaxf(cy - 0.5f * bh, 0.0f), imh);
  float x1 = fminf(fmaxf(cx - 0.5f * bw, 0.0f), imw);
  float y2 = fminf(fmaxf(cy + 0.5f * bh, 0.0f), imh);
  float x2 = fminf(fmaxf(cx + 0.5f * bw, 0.0f), imw);
  bool valid = ((y2 - y1) >= 16.0f) && ((x2 - x1) >= 16.0f);
  float m = fmaxf(s0, s1);
  float e0 = expf(s0 - m), e1 = expf(s1 - m);
  float fg = e1 / (e0 + e1);
  float sc = valid ? fg : -__builtin_inff();

  float* bp = boxws + ((size_t)n * TOTANCH + f) * 4;
  bp[0] = y1; bp[1] = x1; bp[2] = y2; bp[3] = x2;

  unsigned b = __float_as_uint(sc);
  unsigned mono = (b & 0x80000000u) ? ~b : (b | 0x80000000u);
  keys[(size_t)n * SORTN + f] = (((unsigned long long)(~mono)) << 32) | (unsigned long long)(unsigned)f;
}

// ---- bitonic sort step: ascending within each 65536-key segment ----
__global__ void k_bitonic(unsigned long long* __restrict__ keys, int j, int k) {
  int g = blockIdx.x * blockDim.x + threadIdx.x;  // 0 .. 2*65536-1
  int seg = g >> 16;
  int i   = g & (SORTN - 1);
  int ixj = i ^ j;
  if (ixj > i) {
    unsigned long long* base = keys + (size_t)seg * SORTN;
    unsigned long long a = base[i];
    unsigned long long b = base[ixj];
    bool up = ((i & k) == 0);
    if ((up && a > b) || (!up && a < b)) { base[i] = b; base[ixj] = a; }
  }
}

// ---- greedy NMS, exact reference semantics; one block per batch ----
__global__ void __launch_bounds__(1024) k_nms(const unsigned long long* __restrict__ keys,
                                              const float* __restrict__ boxws,
                                              float* __restrict__ topbox,
                                              float* __restrict__ out) {
  __shared__ unsigned okw[(NPRE + 31) / 32];
  __shared__ unsigned keepw[(NPRE + 31) / 32];
  __shared__ int sel[NPOST];
  __shared__ int scnt;
  __shared__ int sflag;
  const int n = blockIdx.x;
  const int tid = threadIdx.x;
  const int bd = blockDim.x;

  for (int i = tid; i < (NPRE + 31) / 32; i += bd) { okw[i] = 0u; keepw[i] = 0u; }
  if (tid == 0) { scnt = 0; sflag = 0; }
  __syncthreads();

  float* tb = topbox + (size_t)n * NPRE * 4;
  for (int i = tid; i < NPRE; i += bd) {
    unsigned long long key = keys[(size_t)n * SORTN + i];
    unsigned idx  = (unsigned)key;
    unsigned mono = ~((unsigned)(key >> 32));
    unsigned fb   = (mono & 0x80000000u) ? (mono & 0x7FFFFFFFu) : ~mono;
    bool ok = (((fb >> 23) & 0xFFu) != 0xFFu) && (idx < (unsigned)TOTANCH);
    float a0 = 0.f, a1 = 0.f, a2 = 0.f, a3 = 0.f;
    if (idx < (unsigned)TOTANCH) {
      const float* bp = boxws + ((size_t)n * TOTANCH + idx) * 4;
      a0 = bp[0]; a1 = bp[1]; a2 = bp[2]; a3 = bp[3];
    }
    tb[(size_t)i * 4 + 0] = a0;
    tb[(size_t)i * 4 + 1] = a1;
    tb[(size_t)i * 4 + 2] = a2;
    tb[(size_t)i * 4 + 3] = a3;
    if (ok) atomicOr(&okw[i >> 5], 1u << (i & 31));
  }
  __threadfence_block();
  __syncthreads();

  const v4f* tb4 = (const v4f*)tb;
  for (int i = 0; i < NPRE; ++i) {
    v4f bi = tb4[i];
    float iarea = (bi[2] - bi[0]) * (bi[3] - bi[1]);
    bool local = false;
    for (int j = tid; j < i; j += bd) {
      if ((keepw[j >> 5] >> (j & 31)) & 1u) {
        v4f bj = tb4[j];
        float ih = fmaxf(fminf(bi[2], bj[2]) - fmaxf(bi[0], bj[0]), 0.0f);
        float iw = fmaxf(fminf(bi[3], bj[3]) - fmaxf(bi[1], bj[1]), 0.0f);
        float inter = ih * iw;
        float jarea = (bj[2] - bj[0]) * (bj[3] - bj[1]);
        float iou = inter / (iarea + jarea - inter + 1e-9f);
        if (iou > NMS_THR) local = true;
      }
    }
    if (local) atomicOr(&sflag, 1);
    __syncthreads();
    if (tid == 0) {
      bool oki = (okw[i >> 5] >> (i & 31)) & 1u;
      if (oki && sflag == 0) {
        keepw[i >> 5] |= (1u << (i & 31));
        if (scnt < NPOST) sel[scnt] = i;
        scnt = scnt + 1;
      }
      sflag = 0;
    }
    __syncthreads();
  }

  int cnt = scnt;
  if (cnt > NPOST) cnt = NPOST;
  for (int t = tid; t < NPOST; t += bd) {
    size_t ro = ((size_t)n * NPOST + t) * 4;
    if (t < cnt) {
      v4f b = tb4[sel[t]];
      out[OFF_ROIS + ro + 0] = b[0];
      out[OFF_ROIS + ro + 1] = b[1];
      out[OFF_ROIS + ro + 2] = b[2];
      out[OFF_ROIS + ro + 3] = b[3];
      out[OFF_VMASK + (size_t)n * NPOST + t] = 1.0f;
    } else {
      out[OFF_ROIS + ro + 0] = 0.0f;
      out[OFF_ROIS + ro + 1] = 0.0f;
      out[OFF_ROIS + ro + 2] = 0.0f;
      out[OFF_ROIS + ro + 3] = 0.0f;
      out[OFF_VMASK + (size_t)n * NPOST + t] = 0.0f;
    }
    out[OFF_RIDX + (size_t)n * NPOST + t] = (float)n;
  }
}

extern "C" void kernel_launch(void* const* d_in, const int* in_sizes, int n_in,
                              void* d_out, int out_size, void* d_ws, size_t ws_size,
                              hipStream_t stream) {
  (void)in_sizes; (void)n_in; (void)out_size; (void)ws_size;
  const float* x       = (const float*)d_in[0];
  const float* conv1_w = (const float*)d_in[1];
  const float* conv1_b = (const float*)d_in[2];
  const float* score_w = (const float*)d_in[3];
  const float* score_b = (const float*)d_in[4];
  const float* loc_w   = (const float*)d_in[5];
  const float* loc_b   = (const float*)d_in[6];
  const int*   img_h   = (const int*)d_in[7];
  const int*   img_w   = (const int*)d_in[8];

  char* ws = (char*)d_ws;
  _Float16* xh = (_Float16*)(ws + WS_XH);
  _Float16* wh = (_Float16*)(ws + WS_WH);
  _Float16* hh = (_Float16*)(ws + WS_HH);
  _Float16* wc = (_Float16*)(ws + WS_WC);
  float*    ho = (float*)(ws + WS_HO);
  float*    boxws = (float*)(ws + WS_BOX);
  unsigned long long* keys = (unsigned long long*)(ws + WS_KEYS);
  float*    topbox = (float*)(ws + WS_TOP);
  float*    out = (float*)d_out;

  k_cvt_x  <<<(NBATCH * SPATIAL * CIN) / 256, 256, 0, stream>>>(x, xh);
  k_cvt_w  <<<(COUT * 9 * CIN) / 256,        256, 0, stream>>>(conv1_w, wh);
  k_cvt_wc <<<(64 * CIN) / 256,              256, 0, stream>>>(loc_w, score_w, wc);

  // 4096 waves, 8 waves/block
  k_conv  <<<512, 256, 0, stream>>>(xh, wh, conv1_b, hh);
  // 512 waves, 8 waves/block
  k_heads <<<64, 256, 0, stream>>>(hh, wc, ho);

  k_init_keys <<<(NBATCH * SORTN) / 256, 256, 0, stream>>>(keys);
  k_decode    <<<(NBATCH * TOTANCH) / 256, 256, 0, stream>>>(ho, loc_b, score_b, img_h, img_w,
                                                             out, boxws, keys);
  for (int k = 2; k <= SORTN; k <<= 1)
    for (int j = k >> 1; j >= 1; j >>= 1)
      k_bitonic <<<(NBATCH * SORTN) / 256, 256, 0, stream>>>(keys, j, k);

  k_nms <<<NBATCH, 1024, 0, stream>>>(keys, boxws, topbox, out);
}
